// EfficientNet_imagenet_66194035966285
// MI455X (gfx1250) — 8-bit matrix operand order repaired and checked by reading, compiled, not run on hardware
//
#include <hip/hip_runtime.h>
#include <hip/hip_bf16.h>

#define EPSBN 1e-5f
#define NB 4

typedef __attribute__((ext_vector_type(8))) int v8i;

__device__ __forceinline__ float signf(float x){ return (x>0.f)?1.f:((x<0.f)?-1.f:0.f); }
__device__ __forceinline__ signed char sign8(float x){
  return (x>0.f)?(signed char)1:((x<0.f)?(signed char)-1:(signed char)0);
}

__global__ void k_stem(const float* __restrict__ x, const float* __restrict__ w,
                       float* __restrict__ y, int Hin, int Hout, long total){
  long i = blockIdx.x*(long)blockDim.x + threadIdx.x;
  if(i>=total) return;
  int wo = (int)(i % Hout); long t = i / Hout;
  int ho = (int)(t % Hout); t /= Hout;
  int co = (int)(t % 32);   int n = (int)(t / 32);
  float acc = 0.f;
  #pragma unroll
  for(int ci=0; ci<3; ++ci){
    #pragma unroll
    for(int kh=0; kh<3; ++kh){
      int hi = ho*2 - 1 + kh;
      if(hi<0 || hi>=Hin) continue;
      #pragma unroll
      for(int kw=0; kw<3; ++kw){
        int wi = wo*2 - 1 + kw;
        if(wi<0 || wi>=Hin) continue;
        acc += x[((long)(n*3+ci)*Hin + hi)*Hin + wi] * signf(w[((co*3+ci)*3+kh)*3+kw]);
      }
    }
  }
  y[i] = acc;
}

__global__ void k_bn2d_stats(const float* __restrict__ x, float* __restrict__ mean,
                             float* __restrict__ var, int N, int C, int HW){
  int c = blockIdx.x;
  float s = 0.f, s2 = 0.f;
  long cnt = (long)N*HW;
  for(long i = threadIdx.x; i < cnt; i += blockDim.x){
    long n = i / HW, hw = i - n*HW;
    float v = x[(n*C + c)*HW + hw];
    s += v; s2 += v*v;
  }
  __shared__ float sh[512], sh2[512];
  sh[threadIdx.x] = s; sh2[threadIdx.x] = s2;
  __syncthreads();
  for(int st = blockDim.x>>1; st>0; st>>=1){
    if((int)threadIdx.x < st){
      sh[threadIdx.x]  += sh[threadIdx.x+st];
      sh2[threadIdx.x] += sh2[threadIdx.x+st];
    }
    __syncthreads();
  }
  if(threadIdx.x==0){
    float m = sh[0]/(float)cnt;
    mean[c] = m;
    var[c]  = sh2[0]/(float)cnt - m*m;
  }
}

__global__ void k_bn2d_apply(const float* __restrict__ x, float* __restrict__ y,
                             const float* __restrict__ mean, const float* __restrict__ var,
                             const float* __restrict__ g, const float* __restrict__ b,
                             int C, int HW, long total){
  long i = blockIdx.x*(long)blockDim.x + threadIdx.x;
  if(i>=total) return;
  int c = (int)((i / HW) % C);
  float v = (x[i]-mean[c])*rsqrtf(var[c]+EPSBN)*g[c] + b[c];
  y[i] = fminf(1.f, fmaxf(-1.f, v));
}

__global__ void k_dw(const float* __restrict__ x, const float* __restrict__ w,
                     float* __restrict__ y, int C, int Hin, int Hout, int s, long total){
  long i = blockIdx.x*(long)blockDim.x + threadIdx.x;
  if(i>=total) return;
  int wo = (int)(i % Hout); long t = i / Hout;
  int ho = (int)(t % Hout); t /= Hout;
  int c  = (int)(t % C);    int n = (int)(t / C);
  float acc = 0.f;
  #pragma unroll
  for(int kh=0; kh<3; ++kh){
    int hi = ho*s - 1 + kh;
    if(hi<0 || hi>=Hin) continue;
    #pragma unroll
    for(int kw=0; kw<3; ++kw){
      int wi = wo*s - 1 + kw;
      if(wi<0 || wi>=Hin) continue;
      acc += signf(x[((long)(n*C+c)*Hin + hi)*Hin + wi]) * signf(w[c*9 + kh*3 + kw]);
    }
  }
  y[i] = acc;
}

__global__ void k_packA(const float* __restrict__ w, signed char* __restrict__ ap,
                        int Rows, int K, int KT, long total){
  long tid = blockIdx.x*(long)blockDim.x + threadIdx.x;
  if(tid>=total) return;
  int  byte = (int)(tid & 1023);
  long tile = tid >> 10;
  int  kt = (int)(tile % KT);
  int  mt = (int)(tile / KT);
  int  lane = byte >> 5, j = byte & 31;
  int  m = mt*16 + (lane & 15);
  int  k = kt*64 + ((lane<16)?0:16) + (j&15) + ((j&16)?32:0);
  signed char v = 0;
  if(m < Rows && k < K) v = sign8(w[(long)m*K + k]);
  ap[tid] = v;
}

__global__ void k_packB_act(const float* __restrict__ x, signed char* __restrict__ bp,
                            int C, int HW, long Mpix, int KT, long total){
  long tid = blockIdx.x*(long)blockDim.x + threadIdx.x;
  if(tid>=total) return;
  int  q = (int)(tid & 1023);
  long tile = tid >> 10;
  int  kt = (int)(tile % KT);
  long nt = tile / KT;
  int  kl = q >> 4;
  int  cl = q & 15;
  long col = nt*16 + cl;
  int  k = kt*64 + kl;
  signed char v = 0;
  if(k < C && col < Mpix){
    long n = col / HW, hw = col - n*HW;
    v = sign8(x[(n*C + k)*(long)HW + hw]);
  }
  int half = (kl >> 4) & 1;
  int j    = (kl & 15) + ((kl & 32) ? 16 : 0);
  int lane = (half ? 16 : 0) + cl;
  bp[tile*1024 + lane*32 + j] = v;
}

__global__ void k_packB_mat(const float* __restrict__ x, signed char* __restrict__ bp,
                            int K, int Rows, int KT, long total){
  long tid = blockIdx.x*(long)blockDim.x + threadIdx.x;
  if(tid>=total) return;
  int  q = (int)(tid & 1023);
  long tile = tid >> 10;
  int  kt = (int)(tile % KT);
  long nt = tile / KT;
  int  kl = q >> 4;
  int  cl = q & 15;
  long col = nt*16 + cl;
  int  k = kt*64 + kl;
  signed char v = 0;
  if(k < K && col < Rows) v = sign8(x[col*(long)K + k]);
  int half = (kl >> 4) & 1;
  int j    = (kl & 15) + ((kl & 32) ? 16 : 0);
  int lane = (half ? 16 : 0) + cl;
  bp[tile*1024 + lane*32 + j] = v;
}

__global__ void __launch_bounds__(256)
k_gemm_bin(const signed char* __restrict__ ap, const signed char* __restrict__ bp,
           const float* __restrict__ bias, float* __restrict__ y,
           int KT, int MT, long NG, int Rows, long Cols, int ld, int HW, int mode){
  long wave = (blockIdx.x*(long)blockDim.x + threadIdx.x) >> 5;
  int  lane = threadIdx.x & 31;
  long tiles = (long)MT*NG;
  if(wave >= tiles) return;
  int  mt = (int)(wave % MT);
  long ng = wave / MT;
  long nt0 = ng*NB;

  v8i z = {0,0,0,0,0,0,0,0};
  v8i acc[NB];
  #pragma unroll
  for(int u=0;u<NB;++u) acc[u] = z;

  const signed char* aB = ap + ((long)mt*KT)*1024 + lane*32;
  const signed char* bB = bp + (nt0*KT)*1024 + lane*32;
  const long bStride = (long)KT*1024;

  for(int kt=0; kt<KT; ++kt){
    v8i a = *(const v8i*)(aB + (long)kt*1024);
    #pragma unroll
    for(int u=0;u<NB;++u){
      v8i b = *(const v8i*)(bB + (long)u*bStride + (long)kt*1024);
      acc[u] = __builtin_amdgcn_wmma_i32_16x16x64_iu8(true, a, true, b, acc[u], false, false);
    }
  }

  int rowBase = mt*16 + ((lane < 16) ? 0 : 8);
  #pragma unroll
  for(int u=0;u<NB;++u){
    long col = (nt0+u)*16 + (lane & 15);
    if(col < Cols){
      #pragma unroll
      for(int j=0;j<8;++j){
        int row = rowBase + j;
        if(row < Rows){
          float v = (float)acc[u][j] + (bias ? bias[row] : 0.f);
          if(mode==0){
            long n = col / HW, hw = col - n*HW;
            y[(n*Rows + row)*(long)HW + hw] = v;
          } else {
            y[col*(long)ld + row] = v;
          }
        }
      }
    }
  }
}

__global__ void k_pool(const float* __restrict__ x, float* __restrict__ y, int HW){
  long nc = blockIdx.x;
  float s = 0.f;
  for(int i=threadIdx.x; i<HW; i+=blockDim.x) s += x[nc*HW + i];
  __shared__ float sh[64];
  sh[threadIdx.x] = s; __syncthreads();
  for(int st=32; st>0; st>>=1){
    if((int)threadIdx.x < st) sh[threadIdx.x] += sh[threadIdx.x+st];
    __syncthreads();
  }
  if(threadIdx.x==0) y[nc] = sh[0]/(float)HW;
}

__global__ void k_bn1d(const float* __restrict__ x, float* __restrict__ y,
                       const float* __restrict__ g, const float* __restrict__ b,
                       int N, int F, int mode){
  int f = blockIdx.x*blockDim.x + threadIdx.x;
  if(f>=F) return;
  float s=0.f, s2=0.f;
  for(int n=0;n<N;++n){ float v = x[(long)n*F+f]; s+=v; s2+=v*v; }
  float m = s/(float)N;
  float var = s2/(float)N - m*m;
  float sc = rsqrtf(var+EPSBN)*g[f], sh = b[f];
  for(int n=0;n<N;++n){
    float v = (x[(long)n*F+f]-m)*sc + sh;
    if(mode==0)      v = fminf(1.f, fmaxf(-1.f, v));
    else if(mode==1) v = 1.f/(1.f+expf(-v));
    y[(long)n*F+f] = v;
  }
}

__global__ void k_fc(const float* __restrict__ in, const float* __restrict__ w,
                     const float* __restrict__ b, float* __restrict__ out,
                     int N, int K, int OF){
  int tid = blockIdx.x*blockDim.x + threadIdx.x;
  if(tid >= N*OF) return;
  int n = tid / OF, of = tid % OF;
  float s = b[of];
  for(int k=0;k<K;++k) s += signf(in[(long)n*K+k]) * signf(w[(long)of*K+k]);
  out[tid] = s;
}

__global__ void k_scale(const float* __restrict__ x, const float* __restrict__ se,
                        const float* __restrict__ resid, float* __restrict__ y,
                        int C, int HW, long total){
  long i = blockIdx.x*(long)blockDim.x + threadIdx.x;
  if(i>=total) return;
  long c = (i / HW) % C;
  long n = i / ((long)HW*C);
  float v = x[i]*se[n*C + c];
  if(resid) v += resid[i];
  y[i] = v;
}

__global__ void k_lsm(const float* __restrict__ x, float* __restrict__ out, int F){
  int row = blockIdx.x;
  __shared__ float sh[256];
  float mx = -1e30f;
  for(int i=threadIdx.x;i<F;i+=blockDim.x) mx = fmaxf(mx, x[(long)row*F+i]);
  sh[threadIdx.x]=mx; __syncthreads();
  for(int st=128; st>0; st>>=1){
    if((int)threadIdx.x<st) sh[threadIdx.x]=fmaxf(sh[threadIdx.x], sh[threadIdx.x+st]);
    __syncthreads();
  }
  float M = sh[0]; __syncthreads();
  float s=0.f;
  for(int i=threadIdx.x;i<F;i+=blockDim.x) s += expf(x[(long)row*F+i]-M);
  sh[threadIdx.x]=s; __syncthreads();
  for(int st=128; st>0; st>>=1){
    if((int)threadIdx.x<st) sh[threadIdx.x]+=sh[threadIdx.x+st];
    __syncthreads();
  }
  float lse = M + logf(sh[0]);
  for(int i=threadIdx.x;i<F;i+=blockDim.x) out[(long)row*F+i] = x[(long)row*F+i] - lse;
}

extern "C" void kernel_launch(void* const* d_in, const int* in_sizes, int n_in,
                              void* d_out, int out_size, void* d_ws, size_t ws_size,
                              hipStream_t stream){
  (void)in_sizes; (void)n_in; (void)out_size; (void)ws_size;
  static const int CFGA[16][4] = {
    {32,16,1,1},{16,24,6,2},{24,24,6,1},{24,40,6,2},{40,40,6,1},
    {40,80,6,1},{80,80,6,1},{80,80,6,1},{80,112,6,2},{112,112,6,1},
    {112,112,6,1},{112,192,6,2},{192,192,6,1},{192,192,6,1},{192,192,6,1},{192,320,6,1}};
  const int BATCH = 32;

  int idx = 0;
  auto nx = [&]() -> const float* { return (const float*)d_in[idx++]; };

  const float* xin0  = nx();
  const float* conv1w = nx();
  const float* bn1g = nx();  const float* bn1b = nx();

  struct Blk {
    const float *w1,*g1,*b1,*w2,*g2,*b2,*w3,*g3,*b3;
    const float *grg,*grb;
    const float *s1g,*s1b,*f2w,*f2b,*s2g,*s2b,*f3w,*f3b,*s3g,*s3b;
  };
  Blk blk[16];
  for(int i=0;i<16;i++){
    int ci=CFGA[i][0], co=CFGA[i][1], s=CFGA[i][3];
    bool res = (s==1 && ci==co);
    Blk &B = blk[i];
    B.w1=nx(); B.g1=nx(); B.b1=nx();
    B.w2=nx(); B.g2=nx(); B.b2=nx();
    B.w3=nx(); B.g3=nx(); B.b3=nx();
    if(res){ B.grg=nx(); B.grb=nx(); } else { B.grg=nullptr; B.grb=nullptr; }
    B.s1g=nx(); B.s1b=nx();
    B.f2w=nx(); B.f2b=nx();
    B.s2g=nx(); B.s2b=nx();
    B.f3w=nx(); B.f3b=nx();
    B.s3g=nx(); B.s3b=nx();
  }
  const float* conv2w = nx();
  const float* bn2g = nx();  const float* bn2b = nx();
  const float* bn3g = nx();  const float* bn3b = nx();
  const float* fcw  = nx();  const float* fcb  = nx();
  const float* bn4g = nx();  const float* bn4b = nx();

  const size_t MBy = 1ull<<20;
  char* w8 = (char*)d_ws;
  float* actA = (float*)(w8 + 0);
  float* actB = (float*)(w8 + 64*MBy);
  float* t1   = (float*)(w8 + 128*MBy);
  float* t2   = (float*)(w8 + 288*MBy);
  float* t3   = (float*)(w8 + 352*MBy);
  signed char* bpk = (signed char*)(w8 + 384*MBy);
  signed char* apk = (signed char*)(w8 + 416*MBy);
  float* misc = (float*)(w8 + 418*MBy);
  float* MEAN = misc;
  float* VAR  = misc + 2048;
  float* POOL = misc + 4096;
  float* S1   = misc + 45056;
  float* S2   = misc + 61440;
  float* S3   = misc + 77824;
  float* FCO  = misc + 94208;

  auto cdiv = [](long a, long b){ return (a + b - 1)/b; };

  auto bn2d_inplace = [&](float* buf, const float* g, const float* b, int C, int HW){
    long total = (long)BATCH*C*HW;
    k_bn2d_stats<<<C, 512, 0, stream>>>(buf, MEAN, VAR, BATCH, C, HW);
    k_bn2d_apply<<<(int)cdiv(total,256), 256, 0, stream>>>(buf, buf, MEAN, VAR, g, b, C, HW, total);
  };

  auto pwconv = [&](const float* act, int Cin, int HW, const float* w, int Cout, float* out){
    int  KT = (Cin + 63)/64;
    int  MT = (Cout + 15)/16;
    long Mpix = (long)BATCH*HW;
    long NT = cdiv(Mpix, 16);
    long NG = cdiv(NT, NB);
    long NTpad = NG*NB;
    long aBytes = (long)MT*KT*1024;
    long bBytes = NTpad*KT*1024;
    k_packA<<<(int)cdiv(aBytes,256), 256, 0, stream>>>(w, apk, Cout, Cin, KT, aBytes);
    k_packB_act<<<(int)cdiv(bBytes,256), 256, 0, stream>>>(act, bpk, Cin, HW, Mpix, KT, bBytes);
    long tiles = (long)MT*NG;
    k_gemm_bin<<<(int)cdiv(tiles*32,256), 256, 0, stream>>>(
        apk, bpk, nullptr, out, KT, MT, NG, Cout, Mpix, 0, HW, 0);
  };

  int H = 112;
  {
    long total = (long)BATCH*32*H*H;
    k_stem<<<(int)cdiv(total,256), 256, 0, stream>>>(xin0, conv1w, t2, 224, H, total);
    k_bn2d_stats<<<32, 512, 0, stream>>>(t2, MEAN, VAR, BATCH, 32, H*H);
    k_bn2d_apply<<<(int)cdiv(total,256), 256, 0, stream>>>(t2, actA, MEAN, VAR, bn1g, bn1b, 32, H*H, total);
  }

  float* cur = actA;
  float* oth = actB;

  for(int i=0;i<16;i++){
    int ci=CFGA[i][0], co=CFGA[i][1], e=CFGA[i][2], s=CFGA[i][3];
    int mid = ci*e, sem = co/8;
    bool res = (s==1 && ci==co);
    int Hout = (H + 2 - 3)/s + 1;
    const Blk &B = blk[i];

    pwconv(cur, ci, H*H, B.w1, mid, t1);
    bn2d_inplace(t1, B.g1, B.b1, mid, H*H);

    long dwTot = (long)BATCH*mid*Hout*Hout;
    k_dw<<<(int)cdiv(dwTot,256), 256, 0, stream>>>(t1, B.w2, t2, mid, H, Hout, s, dwTot);
    bn2d_inplace(t2, B.g2, B.b2, mid, Hout*Hout);

    pwconv(t2, mid, Hout*Hout, B.w3, co, t3);
    bn2d_inplace(t3, B.g3, B.b3, co, Hout*Hout);

    k_pool<<<BATCH*co, 64, 0, stream>>>(t3, S1, Hout*Hout);
    k_bn1d<<<(int)cdiv(co,256), 256, 0, stream>>>(S1, S1, B.s1g, B.s1b, BATCH, co, 0);
    k_fc<<<(int)cdiv((long)BATCH*sem,256), 256, 0, stream>>>(S1, B.f2w, B.f2b, S2, BATCH, co, sem);
    k_bn1d<<<(int)cdiv(sem,256), 256, 0, stream>>>(S2, S2, B.s2g, B.s2b, BATCH, sem, 0);
    k_fc<<<(int)cdiv((long)BATCH*co,256), 256, 0, stream>>>(S2, B.f3w, B.f3b, S3, BATCH, sem, co);
    k_bn1d<<<(int)cdiv(co,256), 256, 0, stream>>>(S3, S3, B.s3g, B.s3b, BATCH, co, 1);

    long oTot = (long)BATCH*co*Hout*Hout;
    k_scale<<<(int)cdiv(oTot,256), 256, 0, stream>>>(t3, S3, res ? cur : nullptr, oth, co, Hout*Hout, oTot);
    if(res) bn2d_inplace(oth, B.grg, B.grb, co, Hout*Hout);

    float* tmp = cur; cur = oth; oth = tmp;
    H = Hout;
  }

  pwconv(cur, 320, H*H, conv2w, 1280, t1);
  bn2d_inplace(t1, bn2g, bn2b, 1280, H*H);

  k_pool<<<BATCH*1280, 64, 0, stream>>>(t1, POOL, H*H);
  k_bn1d<<<(int)cdiv(1280,256), 256, 0, stream>>>(POOL, POOL, bn3g, bn3b, BATCH, 1280, 0);

  {
    int  KT = 1280/64;
    int  MT = (1000+15)/16;
    long NT = BATCH/16;
    long NG = cdiv(NT, NB);
    long NTpad = NG*NB;
    long aBytes = (long)MT*KT*1024;
    long bBytes = NTpad*KT*1024;
    k_packA<<<(int)cdiv(aBytes,256), 256, 0, stream>>>(fcw, apk, 1000, 1280, KT, aBytes);
    k_packB_mat<<<(int)cdiv(bBytes,256), 256, 0, stream>>>(POOL, bpk, 1280, BATCH, KT, bBytes);
    long tiles = (long)MT*NG;
    k_gemm_bin<<<(int)cdiv(tiles*32,256), 256, 0, stream>>>(
        apk, bpk, fcb, FCO, KT, MT, NG, 1000, BATCH, 1000, 0, 1);
  }
  k_bn1d<<<(int)cdiv(1000,256), 256, 0, stream>>>(FCO, FCO, bn4g, bn4b, BATCH, 1000, 2);
  k_lsm<<<BATCH, 256, 0, stream>>>(FCO, (float*)d_out, 1000);
}
